// EncoderStack_41446434406979
// MI455X (gfx1250) — matrix operand order repaired and checked by reading, not compiled, not run on hardware
//
#include <hip/hip_runtime.h>

typedef unsigned short ushort_t;
typedef __attribute__((ext_vector_type(16))) __bf16 v16bf;
typedef __attribute__((ext_vector_type(8)))  float  v8f;

#if __has_builtin(__builtin_amdgcn_global_load_async_to_lds_b128) && \
    __has_builtin(__builtin_amdgcn_s_wait_asynccnt)
#define USE_ASYNC_LDS 1
typedef int v4i __attribute__((vector_size(4 * sizeof(int))));
typedef __attribute__((address_space(1))) v4i* gptr_b128;
typedef __attribute__((address_space(3))) v4i* lptr_b128;
#else
#define USE_ASYNC_LDS 0
#endif

// ---------- helpers ----------
__device__ __forceinline__ ushort_t f2bf(float f) {
    union { float f; unsigned u; } x; x.f = f;
    unsigned u = x.u;
    u += 0x7FFFu + ((u >> 16) & 1u);          // round-to-nearest-even
    return (ushort_t)(u >> 16);
}

union FragAB { uint4 q[2]; v16bf v; };

// ---------- f32 -> bf16 conversion (vectorized) ----------
__global__ __launch_bounds__(256) void cvt_f32_bf16(const float* __restrict__ s,
                                                    ushort_t* __restrict__ d, int n) {
    int i = (blockIdx.x * 256 + threadIdx.x) * 4;
    if (i < n) {
        float4 f = *(const float4*)(s + i);
        unsigned lo = (unsigned)f2bf(f.x) | ((unsigned)f2bf(f.y) << 16);
        unsigned hi = (unsigned)f2bf(f.z) | ((unsigned)f2bf(f.w) << 16);
        *(uint2*)(d + i) = make_uint2(lo, hi);
    }
}

// Fused QKV weight, TRANSPOSED: wqkvT[n][d], n = which*1024 + h*64 + k
__global__ __launch_bounds__(256) void pack_qkv_T(const float* __restrict__ Wq,
                                                  const float* __restrict__ Wk,
                                                  const float* __restrict__ Wv,
                                                  ushort_t* __restrict__ out) {
    int i = blockIdx.x * 256 + threadIdx.x;     // over 3072*1024
    int n = i >> 10, d = i & 1023;
    int which = n >> 10, hk = n & 1023;
    int h = hk >> 6, k = hk & 63;
    const float* W = (which == 0) ? Wq : (which == 1) ? Wk : Wv;
    out[i] = f2bf(W[((size_t)h * 1024 + d) * 64 + k]);
}

// Tiled transpose f32 [R][C] -> bf16 [C][R]
__global__ __launch_bounds__(256) void transpose_f32_bf16(const float* __restrict__ src,
                                                          ushort_t* __restrict__ dst,
                                                          int R, int C) {
    __shared__ ushort_t t[32][40];
    int c0 = blockIdx.x * 32, r0 = blockIdx.y * 32;
    int tx = threadIdx.x & 31, ty = threadIdx.x >> 5;   // 32 x 8
#pragma unroll
    for (int i = 0; i < 4; i++)
        t[ty + i * 8][tx] = f2bf(src[(size_t)(r0 + ty + i * 8) * C + c0 + tx]);
    __syncthreads();
#pragma unroll
    for (int i = 0; i < 4; i++)
        dst[(size_t)(c0 + ty + i * 8) * R + r0 + tx] = t[tx][ty + i * 8];
}

// Tiled transpose bf16 [R][C] (row stride src_ld) -> bf16 [C][R] (row stride dst_ld)
__global__ __launch_bounds__(256) void transpose_bf16(const ushort_t* __restrict__ src,
                                                      ushort_t* __restrict__ dst,
                                                      int R, int C, int src_ld, int dst_ld) {
    __shared__ ushort_t t[32][40];
    int c0 = blockIdx.x * 32, r0 = blockIdx.y * 32;
    int tx = threadIdx.x & 31, ty = threadIdx.x >> 5;
#pragma unroll
    for (int i = 0; i < 4; i++)
        t[ty + i * 8][tx] = src[(size_t)(r0 + ty + i * 8) * src_ld + c0 + tx];
    __syncthreads();
#pragma unroll
    for (int i = 0; i < 4; i++)
        dst[(size_t)(c0 + ty + i * 8) * dst_ld + r0 + tx] = t[tx][ty + i * 8];
}

// ---------- row softmax: f32 scores -> bf16 probabilities ----------
__global__ __launch_bounds__(256) void softmax_rows(const float* __restrict__ S,
                                                    ushort_t* __restrict__ P, int n) {
    const int row = blockIdx.x;
    const float* s = S + (size_t)row * n;
    ushort_t*   p  = P + (size_t)row * n;
    __shared__ float red[256];

    float mx = -3.4e38f;
    for (int i = threadIdx.x; i < n; i += 256) mx = fmaxf(mx, s[i]);
    red[threadIdx.x] = mx; __syncthreads();
    for (int st = 128; st > 0; st >>= 1) {
        if (threadIdx.x < st) red[threadIdx.x] = fmaxf(red[threadIdx.x], red[threadIdx.x + st]);
        __syncthreads();
    }
    mx = red[0]; __syncthreads();

    float sum = 0.0f;
    for (int i = threadIdx.x; i < n; i += 256) sum += __expf(s[i] - mx);
    red[threadIdx.x] = sum; __syncthreads();
    for (int st = 128; st > 0; st >>= 1) {
        if (threadIdx.x < st) red[threadIdx.x] += red[threadIdx.x + st];
        __syncthreads();
    }
    float inv = 1.0f / red[0];
    for (int i = threadIdx.x; i < n; i += 256) p[i] = f2bf(__expf(s[i] - mx) * inv);
}

// ---------- bf16 WMMA GEMM: C[M,N] = alpha * A[M,K] @ Bt[N,K]^T (+bias, ReLU) ----------
// A row-major (lda), Bt row-major [N][K] (ldb) i.e. K-contiguous for both operands.
// Block tile 128x128, 256 threads = 8 waves (4x2), wave tile 32x64 = 2x4 wmma tiles.
// Double-buffered LDS fed by GLOBAL_LOAD_ASYNC_TO_LDS_B128 when available.
template<bool OUT_BF16, bool RELU, bool BIAS>
__global__ __launch_bounds__(256) void gemm_bf16_wmma(
    const ushort_t* __restrict__ A, const ushort_t* __restrict__ Bt,
    void* __restrict__ Cp, const float* __restrict__ bias,
    int M, int N, int K, int lda, int ldb, int ldc, float alpha)
{
#if USE_ASYNC_LDS
    __shared__ ushort_t sA[2][128 * 40];
    __shared__ ushort_t sB[2][128 * 40];
#else
    __shared__ ushort_t sA[1][128 * 40];
    __shared__ ushort_t sB[1][128 * 40];
#endif

    const int tid  = threadIdx.x;
    const int lane = tid & 31;
    const int wave = tid >> 5;
    const int wm = wave >> 1;       // 0..3
    const int wn = wave & 1;        // 0..1
    const int lr = lane & 15;
    const int hi = lane >> 4;

    const int m0 = blockIdx.y * 128;
    const int n0 = blockIdx.x * 128;

    const v8f zero = {0,0,0,0,0,0,0,0};
    v8f acc[2][4];
#pragma unroll
    for (int i = 0; i < 2; i++)
#pragma unroll
        for (int j = 0; j < 4; j++) acc[i][j] = zero;

#if USE_ASYNC_LDS
    auto stage_async = [&](int k0, int b) {
#pragma unroll
        for (int i = 0; i < 2; i++) {
            int chunk = tid + i * 256;            // 0..511
            int r  = chunk >> 2;                  // row of 128
            int kc = (chunk & 3) << 3;            // 0,8,16,24
            __builtin_amdgcn_global_load_async_to_lds_b128(
                (gptr_b128)(A + (size_t)(m0 + r) * lda + k0 + kc),
                (lptr_b128)(&sA[b][r * 40 + kc]), 0, 0);
            __builtin_amdgcn_global_load_async_to_lds_b128(
                (gptr_b128)(Bt + (size_t)(n0 + r) * ldb + k0 + kc),
                (lptr_b128)(&sB[b][r * 40 + kc]), 0, 0);
        }
    };

    int buf = 0;
    stage_async(0, 0);
    for (int k0 = 0; k0 < K; k0 += 32) {
        __builtin_amdgcn_s_wait_asynccnt(0);      // my DMA portion of this tile landed
        __syncthreads();                          // everyone's portion landed
        if (k0 + 32 < K) stage_async(k0 + 32, buf ^ 1);

        FragAB af[2];
#pragma unroll
        for (int tm = 0; tm < 2; tm++) {
            int row = wm * 32 + tm * 16 + lr;
            af[tm].q[0] = *(const uint4*)(&sA[buf][row * 40 + hi * 8]);
            af[tm].q[1] = *(const uint4*)(&sA[buf][row * 40 + 16 + hi * 8]);
        }
        FragAB bfr[4];
#pragma unroll
        for (int tn = 0; tn < 4; tn++) {
            int col = wn * 64 + tn * 16 + lr;
            bfr[tn].q[0] = *(const uint4*)(&sB[buf][col * 40 + hi * 8]);
            bfr[tn].q[1] = *(const uint4*)(&sB[buf][col * 40 + 16 + hi * 8]);
        }
#pragma unroll
        for (int tm = 0; tm < 2; tm++)
#pragma unroll
            for (int tn = 0; tn < 4; tn++)
                acc[tm][tn] = __builtin_amdgcn_wmma_f32_16x16x32_bf16(
                    false, af[tm].v, false, bfr[tn].v, (short)0, acc[tm][tn], false, false);
        buf ^= 1;
    }
#else
    for (int k0 = 0; k0 < K; k0 += 32) {
#pragma unroll
        for (int i = 0; i < 2; i++) {
            int chunk = tid + i * 256;
            int r  = chunk >> 2;
            int kc = (chunk & 3) << 3;
            uint4 va = *(const uint4*)(A  + (size_t)(m0 + r) * lda + k0 + kc);
            uint4 vb = *(const uint4*)(Bt + (size_t)(n0 + r) * ldb + k0 + kc);
            *(uint4*)(&sA[0][r * 40 + kc]) = va;
            *(uint4*)(&sB[0][r * 40 + kc]) = vb;
        }
        __syncthreads();

        FragAB af[2];
#pragma unroll
        for (int tm = 0; tm < 2; tm++) {
            int row = wm * 32 + tm * 16 + lr;
            af[tm].q[0] = *(const uint4*)(&sA[0][row * 40 + hi * 8]);
            af[tm].q[1] = *(const uint4*)(&sA[0][row * 40 + 16 + hi * 8]);
        }
        FragAB bfr[4];
#pragma unroll
        for (int tn = 0; tn < 4; tn++) {
            int col = wn * 64 + tn * 16 + lr;
            bfr[tn].q[0] = *(const uint4*)(&sB[0][col * 40 + hi * 8]);
            bfr[tn].q[1] = *(const uint4*)(&sB[0][col * 40 + 16 + hi * 8]);
        }
#pragma unroll
        for (int tm = 0; tm < 2; tm++)
#pragma unroll
            for (int tn = 0; tn < 4; tn++)
                acc[tm][tn] = __builtin_amdgcn_wmma_f32_16x16x32_bf16(
                    false, af[tm].v, false, bfr[tn].v, (short)0, acc[tm][tn], false, false);
        __syncthreads();
    }
#endif

    // epilogue: lanes<16 -> M r, lanes>=16 -> M 8+r ; N = lane%16
#pragma unroll
    for (int tm = 0; tm < 2; tm++) {
#pragma unroll
        for (int tn = 0; tn < 4; tn++) {
            int col = n0 + wn * 64 + tn * 16 + lr;
            float bb = BIAS ? bias[col] : 0.0f;
            int rbase = m0 + wm * 32 + tm * 16 + hi * 8;
#pragma unroll
            for (int r = 0; r < 8; r++) {
                float v = acc[tm][tn][r] * alpha + bb;
                if (RELU) v = fmaxf(v, 0.0f);
                size_t idx = (size_t)(rbase + r) * ldc + col;
                if (OUT_BF16) ((ushort_t*)Cp)[idx] = f2bf(v);
                else          ((float*)Cp)[idx]    = v;
            }
        }
    }
}

// ---------- launcher ----------
extern "C" void kernel_launch(void* const* d_in, const int* in_sizes, int n_in,
                              void* d_out, int out_size, void* d_ws, size_t ws_size,
                              hipStream_t stream) {
    (void)in_sizes; (void)n_in; (void)out_size; (void)ws_size;
    const float* x  = (const float*)d_in[0];
    const float* Wq = (const float*)d_in[1];
    const float* Wk = (const float*)d_in[2];
    const float* Wv = (const float*)d_in[3];
    const float* Wo = (const float*)d_in[4];
    const float* W1 = (const float*)d_in[5];
    const float* b1 = (const float*)d_in[6];
    const float* W2 = (const float*)d_in[7];
    const float* b2 = (const float*)d_in[8];
    float* out = (float*)d_out;

    const int Bb = 4, T = 2048, DM = 1024, FF = 4096;
    const int BT_ = Bb * T;                 // 8192 rows

    char* ws = (char*)d_ws;
    size_t off = 0;
    auto take = [&](size_t bytes) -> char* {
        char* p = ws + off;
        off = (off + bytes + 255) & ~(size_t)255;
        return p;
    };
    ushort_t* xb    = (ushort_t*)take((size_t)BT_ * DM * 2);
    ushort_t* wqkvT = (ushort_t*)take((size_t)3 * DM * DM * 2);   // [3072][1024]
    ushort_t* woT   = (ushort_t*)take((size_t)DM * DM * 2);       // [1024][1024]
    ushort_t* w1T   = (ushort_t*)take((size_t)FF * DM * 2);       // [4096][1024]
    ushort_t* w2T   = (ushort_t*)take((size_t)DM * FF * 2);       // [1024][4096]
    ushort_t* qkv   = (ushort_t*)take((size_t)BT_ * 3 * DM * 2);
    float*    sc    = (float*)   take((size_t)T * T * 4);
    ushort_t* pr    = (ushort_t*)take((size_t)T * T * 2);
    ushort_t* vT    = (ushort_t*)take((size_t)DM * T * 2);        // [1024][2048], per batch
    ushort_t* attn  = (ushort_t*)take((size_t)BT_ * DM * 2);
    ushort_t* proj  = (ushort_t*)take((size_t)BT_ * DM * 2);
    ushort_t* hdn   = (ushort_t*)take((size_t)BT_ * FF * 2);

    // conversions / weight packing (all B operands become [N][K])
    { int n = BT_ * DM; cvt_f32_bf16<<<n / 1024, 256, 0, stream>>>(x, xb, n); }
    { int n = 3 * DM * DM; pack_qkv_T<<<n / 256, 256, 0, stream>>>(Wq, Wk, Wv, wqkvT); }
    transpose_f32_bf16<<<dim3(DM / 32, DM / 32), 256, 0, stream>>>(Wo, woT, DM, DM);
    transpose_f32_bf16<<<dim3(FF / 32, DM / 32), 256, 0, stream>>>(W1, w1T, DM, FF);
    transpose_f32_bf16<<<dim3(DM / 32, FF / 32), 256, 0, stream>>>(W2, w2T, FF, DM);

    // fused QKV projection: [8192,1024] @ [1024,3072]
    gemm_bf16_wmma<true, false, false><<<dim3(3 * DM / 128, BT_ / 128), 256, 0, stream>>>(
        xb, wqkvT, qkv, nullptr, BT_, 3 * DM, DM, DM, DM, 3 * DM, 1.0f);

    // attention, one batch at a time (16.8MB score tile stays in the 192MB L2)
    for (int b = 0; b < Bb; b++) {
        const ushort_t* qb = qkv + (size_t)b * T * 3 * DM;           // Q rows, stride 3*DM
        const ushort_t* kb = qb + DM;                                // K rows (natural [N][K])
        const ushort_t* vb = qb + 2 * DM;                            // V rows
        // scores = (Q @ K^T) / 8
        gemm_bf16_wmma<false, false, false><<<dim3(T / 128, T / 128), 256, 0, stream>>>(
            qb, kb, sc, nullptr, T, T, DM, 3 * DM, 3 * DM, T, 0.125f);
        softmax_rows<<<T, 256, 0, stream>>>(sc, pr, T);
        // V^T: [2048,1024] (stride 3072) -> [1024][2048]
        transpose_bf16<<<dim3(DM / 32, T / 32), 256, 0, stream>>>(vb, vT, T, DM, 3 * DM, T);
        // att = P @ V
        gemm_bf16_wmma<true, false, false><<<dim3(DM / 128, T / 128), 256, 0, stream>>>(
            pr, vT, attn + (size_t)b * T * DM, nullptr, T, DM, T, T, T, DM, 1.0f);
    }

    // output projection
    gemm_bf16_wmma<true, false, false><<<dim3(DM / 128, BT_ / 128), 256, 0, stream>>>(
        attn, woT, proj, nullptr, BT_, DM, DM, DM, DM, DM, 1.0f);
    // FFN up: ReLU(proj @ W1 + b1)
    gemm_bf16_wmma<true, true, true><<<dim3(FF / 128, BT_ / 128), 256, 0, stream>>>(
        proj, w1T, hdn, b1, BT_, FF, DM, DM, DM, FF, 1.0f);
    // FFN down: hdn @ W2 + b2 -> f32 output
    gemm_bf16_wmma<false, false, true><<<dim3(DM / 128, BT_ / 128), 256, 0, stream>>>(
        hdn, w2T, out, b2, BT_, DM, FF, FF, FF, DM, 1.0f);
}
